// DualAttention_15152644620334
// MI455X (gfx1250) — matrix operand order repaired and checked by reading, not compiled, not run on hardware
//
#include <hip/hip_runtime.h>
#include <stdint.h>

#define B_   32
#define S_   1024
#define D_   512
#define H_   8
#define DH_  64
#define QT   128      // q rows per workgroup: 8 waves x 16
#define KT   32       // keys per tile
// exp2-folded constants: p = exp2(score * SCALE * log2(e)); masked -> exp2(-60*log2(e))
#define K2F   0.18033688011112042f   // (1/8) * log2(e)
#define NEG2F (-86.561702453337815f) // -60 * log2(e): exp ~ 8.8e-27 (normal, ~0 in softmax)

typedef __attribute__((ext_vector_type(16))) __bf16 bf16x16;
typedef __attribute__((ext_vector_type(8)))  __bf16 bf16x8;
typedef __attribute__((ext_vector_type(2)))  __bf16 bf16x2;
typedef __attribute__((ext_vector_type(8)))  float  f32x8;

static __device__ __forceinline__ unsigned short f2bf_u(float f) {
  return __builtin_bit_cast(unsigned short, (__bf16)f);   // native bf16 cvt on gfx1250
}
static __device__ __forceinline__ unsigned pkbf(float x, float y) {
  bf16x2 v; v[0] = (__bf16)x; v[1] = (__bf16)y;           // -> v_cvt_pk_bf16_f32
  return __builtin_bit_cast(unsigned, v);
}
static __device__ __forceinline__ bf16x16 cat8(bf16x8 lo, bf16x8 hi) {
  return __builtin_shufflevector(lo, hi, 0,1,2,3,4,5,6,7,8,9,10,11,12,13,14,15);
}
static __device__ __forceinline__ float fexp2(float x) {
#if __has_builtin(__builtin_amdgcn_exp2f)
  return __builtin_amdgcn_exp2f(x);    // raw v_exp_f32 (args are in safe range)
#else
  return exp2f(x);
#endif
}

__global__ __launch_bounds__(256, 1)
void dual_attn_kernel(const float* __restrict__ qg,
                      const float* __restrict__ kg_,
                      const float* __restrict__ v1g,
                      const float* __restrict__ v2g,
                      const int*   __restrict__ cmg,
                      float*       __restrict__ out) {
  // Double-buffered tiles (ping-pong): stage j+1 while computing j.
  // K tile: [buf][key][d-pair] 2xbf16/uint, row stride 36 uints (16B-aligned pad)
  __shared__ __align__(16) unsigned       k_lds[2 * KT * 36];
  // V tiles transposed: [buf][d][key] bf16, row stride 40
  __shared__ __align__(16) unsigned short v1T[2 * DH_ * 40];
  __shared__ __align__(16) unsigned short v2T[2 * DH_ * 40];
  // Per-wave P staging: [wave][row 16][key 32] bf16, row stride 40
  __shared__ __align__(16) unsigned short p_lds[8 * 16 * 40];

  const int t    = threadIdx.x;
  const int wave = t >> 5;
  const int lane = t & 31;
  const int g    = lane >> 4;    // half-wave group
  const int n    = lane & 15;    // column index (B/C/D layouts)
  const int m    = n;            // A-matrix row owned by this lane

  const int bh = blockIdx.y;
  const int b  = bh >> 3;
  const int h  = bh & 7;
  const int qw = blockIdx.x * QT + wave * 16;   // first q row of this wave

  const size_t headoff = ((size_t)b * S_) * D_ + (size_t)h * DH_;
  const float* qp  = qg  + headoff;
  const float* kp  = kg_ + headoff;
  const float* v1p = v1g + headoff;
  const float* v2p = v2g + headoff;
  const int*   cm  = cmg + b * S_;

  // stage one 32-key tile (K + V1 + V2, fp32 -> bf16) into buffer slot s
  auto stage = [&](int j0, int s) {
    unsigned*       kb  = k_lds + s * (KT * 36);
    unsigned short* v1b = v1T   + s * (DH_ * 40);
    unsigned short* v2b = v2T   + s * (DH_ * 40);
    #pragma unroll
    for (int i = 0; i < 4; ++i) {
      int e   = t + i * 256;   // 0..1023
      int row = e >> 5;        // key within tile
      int p   = e & 31;        // d-pair (d = 2p, 2p+1)
      const size_t go = (size_t)(j0 + row) * D_ + 2 * p;
      float2 fk = *(const float2*)(kp + go);
      kb[row * 36 + p] = pkbf(fk.x, fk.y);
      float2 fa = *(const float2*)(v1p + go);
      v1b[(2*p)   * 40 + row] = f2bf_u(fa.x);
      v1b[(2*p+1) * 40 + row] = f2bf_u(fa.y);
      float2 fb = *(const float2*)(v2p + go);
      v2b[(2*p)   * 40 + row] = f2bf_u(fb.x);
      v2b[(2*p+1) * 40 + row] = f2bf_u(fb.y);
    }
  };

  // ---- Q A-fragments: rows qw..qw+15, K(=DH)=64 -> two 16x32 chunks ----
  // A layout (16-bit, 16x32): lane g=0 holds K {0..7,16..23}, g=1 holds K {8..15,24..31}
  bf16x16 aq[2];
  {
    const float* qrow = qp + (size_t)(qw + m) * D_;
    #pragma unroll
    for (int c = 0; c < 2; ++c) {
      float4 f0 = *(const float4*)(qrow + 32*c +      8*g);
      float4 f1 = *(const float4*)(qrow + 32*c +      8*g + 4);
      float4 f2 = *(const float4*)(qrow + 32*c + 16 + 8*g);
      float4 f3 = *(const float4*)(qrow + 32*c + 16 + 8*g + 4);
      bf16x16 a;
      a[0]=(__bf16)f0.x;  a[1]=(__bf16)f0.y;  a[2]=(__bf16)f0.z;  a[3]=(__bf16)f0.w;
      a[4]=(__bf16)f1.x;  a[5]=(__bf16)f1.y;  a[6]=(__bf16)f1.z;  a[7]=(__bf16)f1.w;
      a[8]=(__bf16)f2.x;  a[9]=(__bf16)f2.y;  a[10]=(__bf16)f2.z; a[11]=(__bf16)f2.w;
      a[12]=(__bf16)f3.x; a[13]=(__bf16)f3.y; a[14]=(__bf16)f3.z; a[15]=(__bf16)f3.w;
      aq[c] = a;
    }
  }

  // constant all-ones B matrix (for row-sum WMMA)
  bf16x16 onesB;
  #pragma unroll
  for (int i = 0; i < 16; ++i) onesB[i] = (__bf16)1.0f;

  const f32x8 zero8 = {0.f,0.f,0.f,0.f,0.f,0.f,0.f,0.f};
  f32x8 o1a[4] = {zero8, zero8, zero8, zero8};
  f32x8 o2a[4] = {zero8, zero8, zero8, zero8};
  f32x8 l_acc  = zero8;   // running softmax denominator per row (C layout)

  stage(0, 0);
  __syncthreads();

  for (int j = 0; j < S_ / KT; ++j) {
    const int j0  = j * KT;
    const int buf = j & 1;

    // stage next tile into the other buffer (overlaps with this tile's WMMAs)
    if (j + 1 < S_ / KT) stage(j0 + KT, buf ^ 1);

    // prefetch tile j+2 into cache (global_prefetch_b8)
    if (j + 2 < S_ / KT) {
      const size_t nb = (size_t)(j0 + 2*KT + (t >> 3)) * D_ + (size_t)(t & 7) * 64;
      __builtin_prefetch(kp  + nb, 0, 0);
      __builtin_prefetch(v1p + nb, 0, 0);
      __builtin_prefetch(v2p + nb, 0, 0);
    }

    const unsigned*       kb  = k_lds + buf * (KT * 36);
    const unsigned short* v1b = v1T   + buf * (DH_ * 40);
    const unsigned short* v2b = v2T   + buf * (DH_ * 40);

    // ---- S = Q * K^T : two 16x16 column groups, K-dim 64 = 2 chunks ----
    f32x8 s0 = zero8, s1 = zero8;
    #pragma unroll
    for (int c = 0; c < 2; ++c) {
      // B layout (32x16): lane holds col n, elems e<8 -> Kinner = g*8 + e, e>=8 -> 16 + g*8 + (e-8)
      const unsigned* b0p = &kb[(n)      * 36 + c*16 + g*4];
      const unsigned* b1p = &kb[(16 + n) * 36 + c*16 + g*4];
      bf16x16 b0 = cat8(*(const bf16x8*)(b0p), *(const bf16x8*)(b0p + 8));
      bf16x16 b1 = cat8(*(const bf16x8*)(b1p), *(const bf16x8*)(b1p + 8));
      s0 = __builtin_amdgcn_wmma_f32_16x16x32_bf16(false, aq[c], false, b0, (short)0, s0, false, false);
      s1 = __builtin_amdgcn_wmma_f32_16x16x32_bf16(false, aq[c], false, b1, (short)0, s1, false, false);
    }

    // ---- mask + exp2 (scale and log2e folded; no max subtraction needed) ----
    const int kg0 = j0 + n;
    const int kg1 = kg0 + 16;
    const int cm0 = cm[kg0];
    const int cm1 = cm[kg1];
    float p0[8], p1[8];
    #pragma unroll
    for (int r = 0; r < 8; ++r) {
      const int qrow = qw + r + 8*g;
      float x0 = (cm0 != 0 && kg0 < qrow) ? s0[r] * K2F : NEG2F;
      float x1 = (cm1 != 0 && kg1 < qrow) ? s1[r] * K2F : NEG2F;
      p0[r] = fexp2(x0);
      p1[r] = fexp2(x1);
    }

    // ---- P (C-layout) -> LDS -> A-layout (same-wave DS ops are in-order) ----
    unsigned short* pw = &p_lds[wave * 16 * 40];
    #pragma unroll
    for (int r = 0; r < 8; ++r) {
      const int row = r + 8*g;
      pw[row*40 + n]      = f2bf_u(p0[r]);
      pw[row*40 + 16 + n] = f2bf_u(p1[r]);
    }
    const unsigned short* prp = &p_lds[(wave*16 + m) * 40];
    bf16x16 aP = cat8(*(const bf16x8*)(prp + 16*g),
                      *(const bf16x8*)(prp + 16*g + 8));

    // ---- row-sum via matrix engine: l_acc += P * ones ----
    l_acc = __builtin_amdgcn_wmma_f32_16x16x32_bf16(false, aP, false, onesB, (short)0, l_acc, false, false);

    // ---- O1 += P*V1, O2 += P*V2 : 4 d-chunks of 16 ----
    #pragma unroll
    for (int cd = 0; cd < 4; ++cd) {
      const unsigned short* vb1 = &v1b[(cd*16 + n) * 40 + g*16];
      const unsigned short* vb2 = &v2b[(cd*16 + n) * 40 + g*16];
      bf16x16 bv1 = cat8(*(const bf16x8*)vb1, *(const bf16x8*)(vb1 + 8));
      bf16x16 bv2 = cat8(*(const bf16x8*)vb2, *(const bf16x8*)(vb2 + 8));
      o1a[cd] = __builtin_amdgcn_wmma_f32_16x16x32_bf16(false, aP, false, bv1, (short)0, o1a[cd], false, false);
      o2a[cd] = __builtin_amdgcn_wmma_f32_16x16x32_bf16(false, aP, false, bv2, (short)0, o2a[cd], false, false);
    }

    __syncthreads();   // staged buffer ready / current buffer free for reuse
  }

  // ---- epilogue: normalize and store both outputs ----
  const size_t o2base = (size_t)B_ * S_ * D_;
  #pragma unroll
  for (int r = 0; r < 8; ++r) {
    const int qrow = qw + r + 8*g;
    const float inv = 1.0f / l_acc[r];
    #pragma unroll
    for (int cd = 0; cd < 4; ++cd) {
      const size_t off = headoff + (size_t)qrow * D_ + cd*16 + n;
      out[off]          = o1a[cd][r] * inv;
      out[o2base + off] = o2a[cd][r] * inv;
    }
  }
}

extern "C" void kernel_launch(void* const* d_in, const int* in_sizes, int n_in,
                              void* d_out, int out_size, void* d_ws, size_t ws_size,
                              hipStream_t stream) {
  (void)in_sizes; (void)n_in; (void)out_size; (void)d_ws; (void)ws_size;
  const float* q  = (const float*)d_in[0];
  const float* k  = (const float*)d_in[1];
  const float* v1 = (const float*)d_in[2];
  const float* v2 = (const float*)d_in[3];
  const int*   cm = (const int*)d_in[4];
  float* out = (float*)d_out;
  dim3 grid(S_ / QT, B_ * H_);
  dual_attn_kernel<<<grid, dim3(256), 0, stream>>>(q, k, v1, v2, cm, out);
}
